// CapsuleNeuralNetworkV2_17085379904304
// MI455X (gfx1250) — compile-verified, matrix operand order checked by reading, not run on hardware
//
#include <hip/hip_runtime.h>

// ---------------------------------------------------------------------------
// Types for CDNA5 WMMA (wave32): v_wmma_f32_16x16x32_bf16
// ---------------------------------------------------------------------------
typedef __attribute__((ext_vector_type(16))) __bf16 bf16x16;
typedef __attribute__((ext_vector_type(8)))  __bf16 bf16x8;
typedef __attribute__((ext_vector_type(4)))  __bf16 bf16x4;
typedef __attribute__((ext_vector_type(8)))  float  f32x8;
typedef __attribute__((ext_vector_type(4)))  float  f32x4;

__device__ __forceinline__ f32x8 wmma_bf16(bf16x16 a, bf16x16 b, f32x8 c) {
  // 8 args: (neg_a, A, neg_b, B, c_mod, C, reuse_a, reuse_b)
  return __builtin_amdgcn_wmma_f32_16x16x32_bf16(false, a, false, b, (short)0, c,
                                                 false, false);
}

// 16-bit A/B fragment load per ISA layout: two 16-byte chunks per lane at
// element offsets  k0 + 8*(lane>>4)  and  +16, row = base + (lane&15).
__device__ __forceinline__ bf16x16 load_frag(const __bf16* base, int row0, int k0,
                                             int stride) {
  const int lane = threadIdx.x & 31;
  const int r    = lane & 15;
  const int kk   = k0 + ((lane >> 4) << 3);
  const __bf16* p = base + (size_t)(row0 + r) * stride + kk;
  bf16x8 lo = *(const bf16x8*)p;
  bf16x8 hi = *(const bf16x8*)(p + 16);
  return __builtin_shufflevector(lo, hi, 0,1,2,3,4,5,6,7,8,9,10,11,12,13,14,15);
}

// ---------------------------------------------------------------------------
// Problem constants
// ---------------------------------------------------------------------------
#define BATCH    32768
#define FEAT     784
#define FV       196
#define KP_CAP   224   // 196 padded to multiple of 32 (K)
#define NPAD_CAP 224   // 196 padded to multiple of 32 (N rows, for 2x2 blocking)
#define LDA_CAP  232   // LDS row stride (elements), breaks bank periodicity
#define NP_CAP   7     // 7 n-pairs of 32 cover 224 columns
#define CAP_B    8     // batches per block
#define CAP_R    (CAP_B * 4)   // 32 (b,t) rows per block
#define KP_DEC   800   // 784 padded to multiple of 32 (K)
#define NPAD_DEC 800   // 784 padded to multiple of 32 (N rows)
#define LDA_DEC  808   // LDS row stride for decoder A tile
#define DEC_R    32    // rows per block in decoder GEMM (one m-pair)

// ---------------------------------------------------------------------------
// Weight repack: f32 [srows x scols] -> bf16 [drows x dcols], zero padded
// ---------------------------------------------------------------------------
__global__ void convert_pad_kernel(const float* __restrict__ src, int srows, int scols,
                                   __bf16* __restrict__ dst, int drows, int dcols) {
  int idx = blockIdx.x * blockDim.x + threadIdx.x;
  if (idx >= drows * dcols) return;
  int r = idx / dcols, c = idx % dcols;
  float v = (r < srows && c < scols) ? src[(size_t)r * scols + c] : 0.0f;
  dst[idx] = (__bf16)v;
}

// ---------------------------------------------------------------------------
// One fused capsule iteration.
//   v = h.reshape(B,4,196); q/k/u = v@W{1,2,3}.T + b; probs = softmax(q k^T);
//   h' = probs @ u
// Block owns 8 batch samples (32 rows); safe for in-place src==dst since the
// whole input tile is staged in LDS before any global write.
// ---------------------------------------------------------------------------
__global__ __launch_bounds__(256) void capsule_iter_kernel(
    const float* __restrict__ src, float* __restrict__ dst,
    const __bf16* __restrict__ W1, const __bf16* __restrict__ W2,
    const __bf16* __restrict__ W3,
    const float* __restrict__ b1, const float* __restrict__ b2,
    const float* __restrict__ b3) {
  __shared__ __attribute__((aligned(16))) __bf16 Abf[CAP_R * LDA_CAP];
  __shared__ __attribute__((aligned(16))) __bf16 Qbf[CAP_R * LDA_CAP];
  __shared__ __attribute__((aligned(16))) __bf16 Kbf[CAP_R * LDA_CAP];
  __shared__ __attribute__((aligned(16))) __bf16 Ubf[CAP_R * LDA_CAP];
  __shared__ float probs[CAP_B][16];

  const int tid = threadIdx.x;
  const int rowBase = blockIdx.x * CAP_R;  // global (b,t) row

  // ---- stage A tile: f32 -> bf16 (196 = 49 float4 per row) ----
  for (int idx = tid; idx < CAP_R * 49; idx += 256) {
    int r = idx / 49, c4 = idx % 49;
    f32x4 v = *(const f32x4*)(src + (size_t)(rowBase + r) * FV + c4 * 4);
    __bf16* d = Abf + r * LDA_CAP + c4 * 4;
    d[0] = (__bf16)v[0]; d[1] = (__bf16)v[1];
    d[2] = (__bf16)v[2]; d[3] = (__bf16)v[3];
  }
  // zero-pad K columns [196, 232)
  for (int idx = tid; idx < CAP_R * (LDA_CAP - FV); idx += 256) {
    int r = idx / (LDA_CAP - FV), c = idx % (LDA_CAP - FV);
    Abf[r * LDA_CAP + FV + c] = (__bf16)0.0f;
  }
  __syncthreads();

  // ---- Q/K/U projections, 2x2 register-blocked WMMA ----
  // Tasks: 3 projections x 7 n-pairs; each task = 32x32 output block.
  const int wave = tid >> 5;
  const int lane = tid & 31;
  for (int tt = wave; tt < 3 * NP_CAP; tt += 8) {
    const int p  = tt / NP_CAP;
    const int n0 = (tt % NP_CAP) * 32;
    const __bf16* Wp = (p == 0) ? W1 : ((p == 1) ? W2 : W3);
    const float*  bp = (p == 0) ? b1 : ((p == 1) ? b2 : b3);
    __bf16* Dmat     = (p == 0) ? Qbf : ((p == 1) ? Kbf : Ubf);

    __builtin_prefetch((const void*)(Wp + (size_t)n0 * KP_CAP), 0, 1);
    __builtin_prefetch((const void*)(Wp + (size_t)(n0 + 16) * KP_CAP), 0, 1);

    f32x8 c00 = {}, c01 = {}, c10 = {}, c11 = {};
    for (int k0 = 0; k0 < KP_CAP; k0 += 32) {
      bf16x16 a0 = load_frag(Abf, 0,  k0, LDA_CAP);
      bf16x16 a1 = load_frag(Abf, 16, k0, LDA_CAP);
      bf16x16 b0 = load_frag(Wp, n0,      k0, KP_CAP);
      bf16x16 b1 = load_frag(Wp, n0 + 16, k0, KP_CAP);
      c00 = wmma_bf16(a0, b0, c00);
      c01 = wmma_bf16(a0, b1, c01);
      c10 = wmma_bf16(a1, b0, c10);
      c11 = wmma_bf16(a1, b1, c11);
    }
    const int nA = n0 + (lane & 15);
    const int nB = nA + 16;
    const float biasA = (nA < FV) ? bp[nA] : 0.0f;
    const float biasB = (nB < FV) ? bp[nB] : 0.0f;
    const int rb = ((lane >> 4) << 3);
#pragma unroll
    for (int r = 0; r < 8; ++r) {
      Dmat[(rb + r) * LDA_CAP + nA]        = (__bf16)(c00[r] + biasA);
      Dmat[(rb + r) * LDA_CAP + nB]        = (__bf16)(c01[r] + biasB);
      Dmat[(16 + rb + r) * LDA_CAP + nA]   = (__bf16)(c10[r] + biasA);
      Dmat[(16 + rb + r) * LDA_CAP + nB]   = (__bf16)(c11[r] + biasB);
    }
  }
  __syncthreads();

  // ---- attention scores + softmax: wave w handles batch w, lanes 0..15 ----
  if (lane < 16) {
    const int t = lane >> 2, s = lane & 3;
    const __bf16* qr = Qbf + (wave * 4 + t) * LDA_CAP;
    const __bf16* kr = Kbf + (wave * 4 + s) * LDA_CAP;
    float sum = 0.0f;
    // 25 x 8 elements covers f in [0,200); cols 196..223 are exact zeros
    for (int c = 0; c < 25; ++c) {
      bf16x8 qa = *(const bf16x8*)(qr + c * 8);
      bf16x8 ka = *(const bf16x8*)(kr + c * 8);
#pragma unroll
      for (int j = 0; j < 8; ++j) sum += (float)qa[j] * (float)ka[j];
    }
    float m = sum;
    m = fmaxf(m, __shfl_xor(m, 1, 32));
    m = fmaxf(m, __shfl_xor(m, 2, 32));
    float e = __expf(sum - m);
    float den = e;
    den += __shfl_xor(den, 1, 32);
    den += __shfl_xor(den, 2, 32);
    probs[wave][lane] = e / den;
  }
  __syncthreads();

  // ---- combine h' = probs @ u, write f32 (float4 per chunk) ----
  for (int idx = tid; idx < CAP_R * 49; idx += 256) {
    int r = idx / 49, c4 = idx % 49;
    int bi = r >> 2, t = r & 3;
    const float* pr = &probs[bi][t * 4];
    f32x4 out = {};
#pragma unroll
    for (int s = 0; s < 4; ++s) {
      float p = pr[s];
      bf16x4 u = *(const bf16x4*)(Ubf + (bi * 4 + s) * LDA_CAP + c4 * 4);
      out[0] += p * (float)u[0]; out[1] += p * (float)u[1];
      out[2] += p * (float)u[2]; out[3] += p * (float)u[3];
    }
    *(f32x4*)(dst + (size_t)(rowBase + r) * FV + c4 * 4) = out;
  }
}

// ---------------------------------------------------------------------------
// Generic WMMA GEMM (2x2 register-blocked):  C = act(A @ Bw.T + bias)
//   A:  f32 [32768 x Kvalid], Bw: bf16 [Npad x KP] (rows n-major, zero padded)
//   Npairs tasks of 32x32 output blocks; store guard n < Nstore (C width),
//   bias guard n < Nvalid.
// ---------------------------------------------------------------------------
template <bool RELU>
__global__ __launch_bounds__(256) void gemm_kernel(
    const float* __restrict__ A, int Kvalid,
    const __bf16* __restrict__ Bw,
    const float* __restrict__ bias, int Nvalid, int Nstore, int Npairs,
    float* __restrict__ C, int ldc) {
  __shared__ __attribute__((aligned(16))) __bf16 Abf[DEC_R * LDA_DEC];
  const int tid = threadIdx.x;
  const int rowBase = blockIdx.x * DEC_R;

  const int K4 = Kvalid >> 2;  // 196
  for (int idx = tid; idx < DEC_R * K4; idx += 256) {
    int r = idx / K4, c4 = idx % K4;
    f32x4 v = *(const f32x4*)(A + (size_t)(rowBase + r) * Kvalid + c4 * 4);
    __bf16* d = Abf + r * LDA_DEC + c4 * 4;
    d[0] = (__bf16)v[0]; d[1] = (__bf16)v[1];
    d[2] = (__bf16)v[2]; d[3] = (__bf16)v[3];
  }
  for (int idx = tid; idx < DEC_R * (KP_DEC - 784); idx += 256) {
    int r = idx / (KP_DEC - 784), c = idx % (KP_DEC - 784);
    Abf[r * LDA_DEC + 784 + c] = (__bf16)0.0f;
  }
  __syncthreads();

  const int wave = tid >> 5;
  const int lane = tid & 31;
  for (int tt = wave; tt < Npairs; tt += 8) {
    const int n0 = tt * 32;
    __builtin_prefetch((const void*)(Bw + (size_t)n0 * KP_DEC), 0, 1);
    __builtin_prefetch((const void*)(Bw + (size_t)(n0 + 16) * KP_DEC), 0, 1);

    f32x8 c00 = {}, c01 = {}, c10 = {}, c11 = {};
    for (int k0 = 0; k0 < KP_DEC; k0 += 32) {
      bf16x16 a0 = load_frag(Abf, 0,  k0, LDA_DEC);
      bf16x16 a1 = load_frag(Abf, 16, k0, LDA_DEC);
      bf16x16 b0 = load_frag(Bw, n0,      k0, KP_DEC);
      bf16x16 b1 = load_frag(Bw, n0 + 16, k0, KP_DEC);
      c00 = wmma_bf16(a0, b0, c00);
      c01 = wmma_bf16(a0, b1, c01);
      c10 = wmma_bf16(a1, b0, c10);
      c11 = wmma_bf16(a1, b1, c11);
    }
    const int nA = n0 + (lane & 15);
    const int nB = nA + 16;
    const float bvA = (nA < Nvalid) ? bias[nA] : 0.0f;
    const float bvB = (nB < Nvalid) ? bias[nB] : 0.0f;
    const int rb = ((lane >> 4) << 3);
#pragma unroll
    for (int r = 0; r < 8; ++r) {
      float d00 = c00[r] + bvA, d01 = c01[r] + bvB;
      float d10 = c10[r] + bvA, d11 = c11[r] + bvB;
      if (RELU) {
        d00 = fmaxf(d00, 0.0f); d01 = fmaxf(d01, 0.0f);
        d10 = fmaxf(d10, 0.0f); d11 = fmaxf(d11, 0.0f);
      }
      if (nA < Nstore) {
        C[(size_t)(rowBase + rb + r) * ldc + nA]      = d00;
        C[(size_t)(rowBase + 16 + rb + r) * ldc + nA] = d10;
      }
      if (nB < Nstore) {
        C[(size_t)(rowBase + rb + r) * ldc + nB]      = d01;
        C[(size_t)(rowBase + 16 + rb + r) * ldc + nB] = d11;
      }
    }
  }
}

// ---------------------------------------------------------------------------
// Row softmax over 10 valid columns of a [B x 16] logits buffer
// ---------------------------------------------------------------------------
__global__ void softmax10_kernel(const float* __restrict__ logits,
                                 float* __restrict__ out) {
  int b = blockIdx.x * blockDim.x + threadIdx.x;
  if (b >= BATCH) return;
  float v[10], m = -1e30f;
#pragma unroll
  for (int o = 0; o < 10; ++o) { v[o] = logits[b * 16 + o]; m = fmaxf(m, v[o]); }
  float s = 0.0f;
#pragma unroll
  for (int o = 0; o < 10; ++o) { v[o] = __expf(v[o] - m); s += v[o]; }
  float inv = 1.0f / s;
#pragma unroll
  for (int o = 0; o < 10; ++o) out[(size_t)b * 10 + o] = v[o] * inv;
}

// ---------------------------------------------------------------------------
// Host launcher
// ---------------------------------------------------------------------------
extern "C" void kernel_launch(void* const* d_in, const int* in_sizes, int n_in,
                              void* d_out, int out_size, void* d_ws, size_t ws_size,
                              hipStream_t stream) {
  const float* x   = (const float*)d_in[0];
  const float* W1  = (const float*)d_in[1];  const float* b1 = (const float*)d_in[2];
  const float* W2  = (const float*)d_in[3];  const float* b2 = (const float*)d_in[4];
  const float* W3  = (const float*)d_in[5];  const float* b3 = (const float*)d_in[6];
  const float* Wd1 = (const float*)d_in[7];  const float* bd1 = (const float*)d_in[8];
  const float* Wd2 = (const float*)d_in[9];  const float* bd2 = (const float*)d_in[10];
  const float* Wo  = (const float*)d_in[11]; const float* bo = (const float*)d_in[12];

  char* ws = (char*)d_ws;
  size_t off = 0;
  auto alloc = [&](size_t bytes) -> void* {
    void* p = ws + off;
    off += (bytes + 255) & ~(size_t)255;
    return p;
  };
  float*  hbuf   = (float*)alloc((size_t)BATCH * FEAT * 4);
  float*  ybuf   = (float*)alloc((size_t)BATCH * FEAT * 4);
  float*  logits = (float*)alloc((size_t)BATCH * 16 * 4);
  __bf16* W1b  = (__bf16*)alloc((size_t)NPAD_CAP * KP_CAP * 2);
  __bf16* W2b  = (__bf16*)alloc((size_t)NPAD_CAP * KP_CAP * 2);
  __bf16* W3b  = (__bf16*)alloc((size_t)NPAD_CAP * KP_CAP * 2);
  __bf16* Wd1b = (__bf16*)alloc((size_t)NPAD_DEC * KP_DEC * 2);
  __bf16* Wd2b = (__bf16*)alloc((size_t)NPAD_DEC * KP_DEC * 2);
  __bf16* Wob  = (__bf16*)alloc((size_t)32 * KP_DEC * 2);

  auto cvt = [&](const float* s, int sr, int sc, __bf16* d, int dr, int dc) {
    int total = dr * dc;
    convert_pad_kernel<<<(total + 255) / 256, 256, 0, stream>>>(s, sr, sc, d, dr, dc);
  };
  cvt(W1, FV, FV, W1b, NPAD_CAP, KP_CAP);
  cvt(W2, FV, FV, W2b, NPAD_CAP, KP_CAP);
  cvt(W3, FV, FV, W3b, NPAD_CAP, KP_CAP);
  cvt(Wd1, FEAT, FEAT, Wd1b, NPAD_DEC, KP_DEC);
  cvt(Wd2, FEAT, FEAT, Wd2b, NPAD_DEC, KP_DEC);
  cvt(Wo, 10, FEAT, Wob, 32, KP_DEC);

  // 8 capsule iterations; first reads x, rest update hbuf in place (L2-resident)
  const int capBlocks = BATCH / CAP_B;  // 4096
  capsule_iter_kernel<<<capBlocks, 256, 0, stream>>>(x, hbuf, W1b, W2b, W3b, b1, b2, b3);
  for (int it = 1; it < 8; ++it)
    capsule_iter_kernel<<<capBlocks, 256, 0, stream>>>(hbuf, hbuf, W1b, W2b, W3b,
                                                       b1, b2, b3);

  const int decBlocks = BATCH / DEC_R;  // 1024
  gemm_kernel<true ><<<decBlocks, 256, 0, stream>>>(hbuf, FEAT, Wd1b, bd1,
                                                    FEAT, FEAT, 25, ybuf, FEAT);
  gemm_kernel<false><<<decBlocks, 256, 0, stream>>>(ybuf, FEAT, Wd2b, bd2,
                                                    FEAT, FEAT, 25, hbuf, FEAT);
  gemm_kernel<false><<<decBlocks, 256, 0, stream>>>(hbuf, FEAT, Wob, bo,
                                                    10, 16, 1, logits, 16);
  softmax10_kernel<<<(BATCH + 255) / 256, 256, 0, stream>>>(logits, (float*)d_out);
}
